// CustomResNetMambaPosDCN4D_dcnv3_18897856103316
// MI455X (gfx1250) — matrix operand order repaired and checked by reading, not compiled, not run on hardware
//
#include <hip/hip_runtime.h>
#include <cstddef>

// ---------------------------------------------------------------------------
// Types for CDNA5 WMMA (gfx1250, wave32)
// ---------------------------------------------------------------------------
typedef __attribute__((ext_vector_type(16))) __bf16 v16bf;
typedef __attribute__((ext_vector_type(8)))  __bf16 v8bf;
typedef __attribute__((ext_vector_type(8)))  float  v8f;

// Problem constants (fixed by the reference)
#define NB   8
#define HH   56
#define WW   56
#define CC   256
#define MROWS (NB * HH * WW)     // 25088
#define GG   8
#define GC   32
#define PP   9
#define HID  1024

// GEMM tiling
#define BM 128
#define BN 128
#define BK 32
#define LSTR 40   // LDS row stride in bf16 elements (80 B, 16B-aligned)

__device__ __forceinline__ float gelu_exact(float x) {
  return 0.5f * x * (1.0f + erff(x * 0.70710678118654752f));
}

// ---------------------------------------------------------------------------
// WMMA GEMM:  C[M,Ntot] = A[M,K](bf16) * Bt[Ntot,K]^T(bf16)  + epilogue
// MODE 0: plain A.  MODE 1: conv3x3 implicit im2col A from NHWC bf16 image,
//         K = 9*256, tap-major (tap = ky*3+kx).
// Software-pipelined: global loads for tile k+1 are issued before the WMMA
// compute of tile k, hiding HBM/L2 latency behind the matrix pipe.
// ---------------------------------------------------------------------------
struct EpiParams {
  const float* bias;                       // nullable
  const float* bn_g; const float* bn_b;    // epi==2 only
  const float* bn_m; const float* bn_v;
  float*  out_f32;                         // nullable
  __bf16* out_bf16;                        // nullable
  int epi;                                 // 0: +bias, 1: gelu(+bias), 2: BN+ReLU
};

template <int MODE>
__device__ __forceinline__ void load_tiles(const __bf16* __restrict__ A,
                                           const __bf16* __restrict__ Bt,
                                           int M, int Ntot, int K, int k0,
                                           int m0, int n0, int frow, int fcol,
                                           v8bf& a0, v8bf& a1, v8bf& b0, v8bf& b1) {
  v8bf z = {};
  a0 = z; a1 = z; b0 = z; b1 = z;
  if (MODE == 0) {
    const int r = m0 + frow;
    if (r < M) {
      const __bf16* p = A + (size_t)r * K + (size_t)(k0 + fcol);
      a0 = *(const v8bf*)p;
      a1 = *(const v8bf*)(p + 8);
      if (k0 + 2 * BK < K)  // hint tile k+2 into L2 (global_prefetch_b8)
        __builtin_prefetch((const void*)(p + 2 * BK), 0, 1);
    }
  } else {
    // conv3x3 implicit: A is the NHWC bf16 image, K = 9*CC (tap-major)
    const int r = m0 + frow;
    if (r < M) {
      const int tap = k0 >> 8;                 // 0..8 (CC==256, BK|CC)
      const int c0  = (k0 & (CC - 1)) + fcol;
      const int ky  = tap / 3 - 1, kx = tap % 3 - 1;
      const int n   = r / (HH * WW);
      const int hw  = r % (HH * WW);
      const int h   = hw / WW + ky, w = hw % WW + kx;
      if (h >= 0 && h < HH && w >= 0 && w < WW) {
        const __bf16* p = A + ((((size_t)n * HH + h) * WW + w) * CC + c0);
        a0 = *(const v8bf*)p;
        a1 = *(const v8bf*)(p + 8);
      }
    }
  }
  {
    const int r = n0 + frow;
    if (r < Ntot) {
      const __bf16* p = Bt + (size_t)r * K + (size_t)(k0 + fcol);
      b0 = *(const v8bf*)p;
      b1 = *(const v8bf*)(p + 8);
    }
  }
}

template <int MODE>
__global__ __launch_bounds__(256)
void k_gemm_wmma(const __bf16* __restrict__ A, const __bf16* __restrict__ Bt,
                 int M, int Ntot, int K, EpiParams ep) {
  __shared__ __bf16 As[BM * LSTR];
  __shared__ __bf16 Bs[BN * LSTR];

  const int tid  = threadIdx.x;
  const int lane = tid & 31;
  const int wid  = tid >> 5;      // 0..7
  const int wm   = wid >> 1;      // 0..3  -> 4 waves along M
  const int wn   = wid & 1;       // 0..1  -> 2 waves along N
  const int hlf  = lane >> 4;     // lane-half (ISA fragment layout)
  const int l16  = lane & 15;

  const int m0 = blockIdx.y * BM;
  const int n0 = blockIdx.x * BN;

  // LDS fill mapping: 256 threads, each moves 16 bf16 of one row
  const int frow = tid >> 1;          // 0..127
  const int fcol = (tid & 1) * 16;    // 0 or 16

  v8f zero8f = {};
  v8f acc[2][4];
#pragma unroll
  for (int i = 0; i < 2; ++i)
#pragma unroll
    for (int j = 0; j < 4; ++j) acc[i][j] = zero8f;

  // Prologue: stage tile 0 into registers
  v8bf a0, a1, b0, b1;
  load_tiles<MODE>(A, Bt, M, Ntot, K, 0, m0, n0, frow, fcol, a0, a1, b0, b1);

  for (int k0 = 0; k0 < K; k0 += BK) {
    __syncthreads();   // previous compute done; LDS free
    *(v8bf*)&As[frow * LSTR + fcol]     = a0;
    *(v8bf*)&As[frow * LSTR + fcol + 8] = a1;
    *(v8bf*)&Bs[frow * LSTR + fcol]     = b0;
    *(v8bf*)&Bs[frow * LSTR + fcol + 8] = b1;
    __syncthreads();

    // Issue next tile's global loads BEFORE compute (latency hiding)
    if (k0 + BK < K)
      load_tiles<MODE>(A, Bt, M, Ntot, K, k0 + BK, m0, n0, frow, fcol,
                       a0, a1, b0, b1);

    // A fragment (16x32 bf16): lane<16 -> K{0..7,16..23}, lane>=16 -> K{8..15,24..31}
    v16bf afrag[2];
#pragma unroll
    for (int fm = 0; fm < 2; ++fm) {
      const int r = wm * 32 + fm * 16 + l16;
      union { v16bf v; v8bf h[2]; } u;
      u.h[0] = *(const v8bf*)&As[r * LSTR + hlf * 8];
      u.h[1] = *(const v8bf*)&As[r * LSTR + hlf * 8 + 16];
      afrag[fm] = u.v;
    }
    // B fragment (32x16): lane<16 -> K{0..7,16..23}, lane>=16 -> K{8..15,24..31} (column = l16)
    v16bf bfrag[4];
#pragma unroll
    for (int fn = 0; fn < 4; ++fn) {
      const int r = wn * 64 + fn * 16 + l16;
      union { v16bf v; v8bf h[2]; } u;
      u.h[0] = *(const v8bf*)&Bs[r * LSTR + hlf * 8];
      u.h[1] = *(const v8bf*)&Bs[r * LSTR + hlf * 8 + 16];
      bfrag[fn] = u.v;
    }

#pragma unroll
    for (int fm = 0; fm < 2; ++fm)
#pragma unroll
      for (int fn = 0; fn < 4; ++fn)
        acc[fm][fn] = __builtin_amdgcn_wmma_f32_16x16x32_bf16(
            false, afrag[fm], false, bfrag[fn], (short)0, acc[fm][fn],
            false, false);
  }

  // Epilogue. C/D layout: lanes 0-15 hold M rows j (VGPR j), lanes 16-31 rows 8+j.
#pragma unroll
  for (int fm = 0; fm < 2; ++fm) {
#pragma unroll
    for (int fn = 0; fn < 4; ++fn) {
      const int cc = n0 + wn * 64 + fn * 16 + l16;
      if (cc >= Ntot) continue;
      const int rbase = m0 + wm * 32 + fm * 16 + hlf * 8;
      float bias = 0.f, scale = 1.f, shift = 0.f;
      if (ep.epi == 2) {
        const float s = rsqrtf(ep.bn_v[cc] + 1e-5f) * ep.bn_g[cc];
        scale = s;
        shift = ep.bn_b[cc] - ep.bn_m[cc] * s;
      } else if (ep.bias) {
        bias = ep.bias[cc];
      }
#pragma unroll
      for (int j = 0; j < 8; ++j) {
        const int r = rbase + j;
        if (r >= M) continue;
        float v = acc[fm][fn][j];
        if (ep.epi == 2)      { v = fmaxf(v * scale + shift, 0.f); }
        else                  { v += bias; if (ep.epi == 1) v = gelu_exact(v); }
        const size_t o = (size_t)r * Ntot + cc;
        if (ep.out_f32)  ep.out_f32[o]  = v;
        if (ep.out_bf16) ep.out_bf16[o] = (__bf16)v;
      }
    }
  }
}

// ---------------------------------------------------------------------------
// Elementwise / prep kernels
// ---------------------------------------------------------------------------
__global__ void k_nchw_to_nhwc_bf16(const float* __restrict__ src,
                                    __bf16* __restrict__ dst, int total) {
  const int idx = blockIdx.x * 256 + threadIdx.x;
  if (idx >= total) return;
  const int c  = idx & (CC - 1);
  const int t  = idx >> 8;
  const int hw = t % (HH * WW);
  const int n  = t / (HH * WW);
  dst[idx] = (__bf16)src[((size_t)n * CC + c) * (HH * WW) + hw];
}

__global__ void k_transpose_bf16(const float* __restrict__ src,  // [K][N]
                                 __bf16* __restrict__ dst,       // [N][K]
                                 int K, int N, int total) {
  const int idx = blockIdx.x * 256 + threadIdx.x;
  if (idx >= total) return;
  const int k = idx / N, n = idx % N;
  dst[(size_t)n * K + k] = (__bf16)src[idx];
}

__global__ void k_prep_conv1(const float* __restrict__ src,  // OIHW (256,256,3,3)
                             __bf16* __restrict__ dst,       // [O][tap*256+c]
                             int total) {
  const int idx = blockIdx.x * 256 + threadIdx.x;
  if (idx >= total) return;
  const int o = idx / (9 * CC), k = idx % (9 * CC);
  const int tap = k >> 8, c = k & (CC - 1);
  const int ky = tap / 3, kx = tap % 3;
  dst[idx] = (__bf16)src[(((size_t)o * CC + c) * 3 + ky) * 3 + kx];
}

// 256-thread block sum reduction
__device__ __forceinline__ float blk_sum(float v, float* sb) {
  const int t = threadIdx.x;
  sb[t] = v;
  __syncthreads();
#pragma unroll
  for (int s = 128; s >= 1; s >>= 1) {
    if (t < s) sb[t] += sb[t + s];
    __syncthreads();
  }
  const float r = sb[0];
  __syncthreads();
  return r;
}

// depthwise 3x3 (+bias) -> LayerNorm(eps 1e-6) -> GELU -> bf16
__global__ __launch_bounds__(256)
void k_dw_ln_gelu(const float* __restrict__ t0, const float* __restrict__ dw_w,
                  const float* __restrict__ dw_b, const float* __restrict__ g,
                  const float* __restrict__ b, __bf16* __restrict__ x1h) {
  __shared__ float sb[256];
  const int m = blockIdx.x, c = threadIdx.x;
  const int n = m / (HH * WW), hw = m % (HH * WW);
  const int h = hw / WW, w = hw % WW;
  float acc = dw_b[c];
#pragma unroll
  for (int ky = 0; ky < 3; ++ky) {
#pragma unroll
    for (int kx = 0; kx < 3; ++kx) {
      const int hh = h + ky - 1, ww2 = w + kx - 1;
      if (hh >= 0 && hh < HH && ww2 >= 0 && ww2 < WW)
        acc += t0[(((size_t)n * HH + hh) * WW + ww2) * CC + c] *
               dw_w[(ky * 3 + kx) * CC + c];
    }
  }
  const float mu  = blk_sum(acc, sb) * (1.0f / CC);
  const float dv  = acc - mu;
  const float var = blk_sum(dv * dv, sb) * (1.0f / CC);
  float y = dv * rsqrtf(var + 1e-6f) * g[c] + b[c];
  x1h[(size_t)m * CC + c] = (__bf16)gelu_exact(y);
}

__global__ void k_softmax9(float* __restrict__ p, int total) {
  const int i = blockIdx.x * 256 + threadIdx.x;
  if (i >= total) return;
  float* q = p + (size_t)i * PP;
  float v[PP], mx = -1e30f;
#pragma unroll
  for (int j = 0; j < PP; ++j) { v[j] = q[j]; mx = fmaxf(mx, v[j]); }
  float s = 0.f;
#pragma unroll
  for (int j = 0; j < PP; ++j) { v[j] = expf(v[j] - mx); s += v[j]; }
  const float inv = 1.0f / s;
#pragma unroll
  for (int j = 0; j < PP; ++j) q[j] = v[j] * inv;
}

__device__ __forceinline__ float dcn_fetch(const float* __restrict__ xp,
                                           int n, int iy, int ix, int c) {
  // padded coords; valid image region is [1,56]x[1,56] of the 58x58 pad frame
  if (ix < 1 || ix > WW || iy < 1 || iy > HH) return 0.f;
  return xp[(((size_t)n * HH + (iy - 1)) * WW + (ix - 1)) * CC + c];
}

// DCNv3 bilinear sampling + mask-weighted sum -> bf16 (M x 256)
__global__ __launch_bounds__(256)
void k_dcn_sample(const float* __restrict__ xproj, const float* __restrict__ offs,
                  const float* __restrict__ msk, __bf16* __restrict__ outh) {
  const int m = blockIdx.x, c = threadIdx.x;
  const int g = c >> 5;
  const int n = m / (HH * WW), hw = m % (HH * WW);
  const int h = hw / WW, w = hw % WW;
  const float* off = offs + (size_t)m * (GG * PP * 2) + g * (PP * 2);
  const float* mp  = msk  + (size_t)m * (GG * PP)     + g * PP;
  float acc = 0.f;
#pragma unroll
  for (int p = 0; p < PP; ++p) {
    // gx = w + (p/3) + off_x  (padded coords); gy = h + (p%3) + off_y
    const float gx = (float)(w + p / 3) + off[p * 2 + 0];
    const float gy = (float)(h + p % 3) + off[p * 2 + 1];
    const float x0f = floorf(gx), y0f = floorf(gy);
    const float wx = gx - x0f, wy = gy - y0f;
    const int x0 = (int)x0f, y0 = (int)y0f;
    const float v00 = dcn_fetch(xproj, n, y0,     x0,     c);
    const float v10 = dcn_fetch(xproj, n, y0,     x0 + 1, c);
    const float v01 = dcn_fetch(xproj, n, y0 + 1, x0,     c);
    const float v11 = dcn_fetch(xproj, n, y0 + 1, x0 + 1, c);
    const float s = v00 * (1.f - wx) * (1.f - wy) + v10 * wx * (1.f - wy) +
                    v01 * (1.f - wx) * wy         + v11 * wx * wy;
    acc += s * mp[p];
  }
  outh[(size_t)m * CC + c] = (__bf16)acc;
}

// out = base + gamma * LayerNorm(d; g,b; eps 1e-5); optional bf16 copy
__global__ __launch_bounds__(256)
void k_ln_scale_residual(const float* __restrict__ base, const float* __restrict__ d,
                         const float* __restrict__ g, const float* __restrict__ b,
                         const float* __restrict__ gamma,
                         float* __restrict__ out, __bf16* __restrict__ outh) {
  __shared__ float sb[256];
  const int m = blockIdx.x, c = threadIdx.x;
  const size_t o = (size_t)m * CC + c;
  const float v  = d[o];
  const float mu  = blk_sum(v, sb) * (1.0f / CC);
  const float dv  = v - mu;
  const float var = blk_sum(dv * dv, sb) * (1.0f / CC);
  const float y = dv * rsqrtf(var + 1e-5f) * g[c] + b[c];
  const float r = base[o] + gamma[c] * y;
  out[o] = r;
  if (outh) outh[o] = (__bf16)r;
}

// final: relu(BN(t2 NHWC -> NCHW) + x)
__global__ void k_bn_add_relu(const float* __restrict__ t2, const float* __restrict__ x,
                              const float* __restrict__ g, const float* __restrict__ b,
                              const float* __restrict__ mean, const float* __restrict__ var,
                              float* __restrict__ out, int total) {
  const int idx = blockIdx.x * 256 + threadIdx.x;
  if (idx >= total) return;
  const int hw = idx % (HH * WW);
  const int c  = (idx / (HH * WW)) & (CC - 1);
  const int n  = idx / ((HH * WW) * CC);
  const float v = t2[((size_t)n * (HH * WW) + hw) * CC + c];
  const float s = rsqrtf(var[c] + 1e-5f);
  const float r = (v - mean[c]) * (g[c] * s) + b[c] + x[idx];
  out[idx] = fmaxf(r, 0.f);
}

// ---------------------------------------------------------------------------
// Host launcher
// ---------------------------------------------------------------------------
extern "C" void kernel_launch(void* const* d_in, const int* in_sizes, int n_in,
                              void* d_out, int out_size, void* d_ws, size_t ws_size,
                              hipStream_t stream) {
  (void)in_sizes; (void)n_in; (void)out_size; (void)ws_size;

  const float* x       = (const float*)d_in[0];
  const float* conv1_w = (const float*)d_in[1];
  const float* bn1_g   = (const float*)d_in[2];
  const float* bn1_b   = (const float*)d_in[3];
  const float* bn1_m   = (const float*)d_in[4];
  const float* bn1_v   = (const float*)d_in[5];
  const float* ln1_g   = (const float*)d_in[6];
  const float* ln1_b   = (const float*)d_in[7];
  const float* dw_w    = (const float*)d_in[8];
  const float* dw_b    = (const float*)d_in[9];
  const float* dwln_g  = (const float*)d_in[10];
  const float* dwln_b  = (const float*)d_in[11];
  const float* off_w   = (const float*)d_in[12];
  const float* off_b   = (const float*)d_in[13];
  const float* msk_w   = (const float*)d_in[14];
  const float* msk_b   = (const float*)d_in[15];
  const float* inp_w   = (const float*)d_in[16];
  const float* inp_b   = (const float*)d_in[17];
  const float* outp_w  = (const float*)d_in[18];
  const float* outp_b  = (const float*)d_in[19];
  const float* gamma1  = (const float*)d_in[20];
  const float* ln2_g   = (const float*)d_in[21];
  const float* ln2_b   = (const float*)d_in[22];
  const float* fc1_w   = (const float*)d_in[23];
  const float* fc1_b   = (const float*)d_in[24];
  const float* fc2_w   = (const float*)d_in[25];
  const float* fc2_b   = (const float*)d_in[26];
  const float* gamma2  = (const float*)d_in[27];
  const float* bn2_g   = (const float*)d_in[28];
  const float* bn2_b   = (const float*)d_in[29];
  const float* bn2_m   = (const float*)d_in[30];
  const float* bn2_v   = (const float*)d_in[31];

  char* ws = (char*)d_ws;
  // weight slots (bf16, pre-transposed [N][K])
  const size_t o_w1t   = 0;                               // 256 x 2304
  const size_t o_inpwt = o_w1t   + (size_t)256 * 2304 * 2;
  const size_t o_offwt = o_inpwt + (size_t)256 * 256 * 2;
  const size_t o_mskwt = o_offwt + (size_t)144 * 256 * 2;
  const size_t o_outwt = o_mskwt + (size_t)72  * 256 * 2;
  const size_t o_fc1wt = o_outwt + (size_t)256 * 256 * 2;
  const size_t o_fc2wt = o_fc1wt + (size_t)1024 * 256 * 2;
  // activation slots with lifetime reuse
  const size_t o_B = o_fc2wt + (size_t)256 * 1024 * 2;    // t0 -> hout      (f32 MxC)
  const size_t o_C = o_B + (size_t)MROWS * CC * 4;        // xproj -> t1     (f32 MxC)
  const size_t o_D = o_C + (size_t)MROWS * CC * 4;        // xh/x1h -> dproj -> t2
  const size_t o_E = o_D + (size_t)MROWS * CC * 4;        // t0h -> masks
  const size_t o_F = o_E + (size_t)MROWS * CC * 2;        // offsets -> t1h
  const size_t o_G = o_F + (size_t)MROWS * 144 * 4;       // dcnouth -> hh

  __bf16* xh      = (__bf16*)(ws + o_D);
  __bf16* w1t     = (__bf16*)(ws + o_w1t);
  __bf16* inpwt   = (__bf16*)(ws + o_inpwt);
  __bf16* offwt   = (__bf16*)(ws + o_offwt);
  __bf16* mskwt   = (__bf16*)(ws + o_mskwt);
  __bf16* outwt   = (__bf16*)(ws + o_outwt);
  __bf16* fc1wt   = (__bf16*)(ws + o_fc1wt);
  __bf16* fc2wt   = (__bf16*)(ws + o_fc2wt);
  float*  t0      = (float*)(ws + o_B);
  __bf16* t0h     = (__bf16*)(ws + o_E);
  float*  xproj   = (float*)(ws + o_C);
  __bf16* x1h     = (__bf16*)(ws + o_D);
  float*  offsets = (float*)(ws + o_F);
  float*  masks   = (float*)(ws + o_E);
  __bf16* dcnouth = (__bf16*)(ws + o_G);
  float*  dproj   = (float*)(ws + o_D);
  float*  t1      = (float*)(ws + o_C);
  __bf16* t1h     = (__bf16*)(ws + o_F);
  __bf16* hh      = (__bf16*)(ws + o_G);
  float*  hout    = (float*)(ws + o_B);
  float*  t2      = (float*)(ws + o_D);

  const int total_img = NB * CC * HH * WW;      // 6,422,528
  const dim3 blk(256);

  // 1) NCHW f32 -> NHWC bf16
  k_nchw_to_nhwc_bf16<<<(total_img + 255) / 256, blk, 0, stream>>>(x, xh, total_img);

  // 2) weight prep (bf16, transposed)
  k_prep_conv1<<<(256 * 2304 + 255) / 256, blk, 0, stream>>>(conv1_w, w1t, 256 * 2304);
  k_transpose_bf16<<<(256 * 256 + 255) / 256, blk, 0, stream>>>(inp_w, inpwt, 256, 256, 256 * 256);
  k_transpose_bf16<<<(256 * 144 + 255) / 256, blk, 0, stream>>>(off_w, offwt, 256, 144, 256 * 144);
  k_transpose_bf16<<<(256 * 72 + 255) / 256, blk, 0, stream>>>(msk_w, mskwt, 256, 72, 256 * 72);
  k_transpose_bf16<<<(256 * 256 + 255) / 256, blk, 0, stream>>>(outp_w, outwt, 256, 256, 256 * 256);
  k_transpose_bf16<<<(256 * 1024 + 255) / 256, blk, 0, stream>>>(fc1_w, fc1wt, 256, 1024, 256 * 1024);
  k_transpose_bf16<<<(1024 * 256 + 255) / 256, blk, 0, stream>>>(fc2_w, fc2wt, 1024, 256, 1024 * 256);

  const dim3 gN256(2, MROWS / BM);   // Ntot=256
  const dim3 gN144(2, MROWS / BM);
  const dim3 gN72(1, MROWS / BM);
  const dim3 gN1024(8, MROWS / BM);

  // 3) conv1 (implicit GEMM) + BN + ReLU -> t0 (f32) + t0h (bf16)
  {
    EpiParams ep{nullptr, bn1_g, bn1_b, bn1_m, bn1_v, t0, t0h, 2};
    k_gemm_wmma<1><<<gN256, blk, 0, stream>>>(xh, w1t, MROWS, 256, 9 * CC, ep);
  }
  // 4) input_proj -> xproj (f32)
  {
    EpiParams ep{inp_b, nullptr, nullptr, nullptr, nullptr, xproj, nullptr, 0};
    k_gemm_wmma<0><<<gN256, blk, 0, stream>>>(t0h, inpwt, MROWS, 256, 256, ep);
  }
  // 5) depthwise + LN + GELU -> x1h (bf16)
  k_dw_ln_gelu<<<MROWS, blk, 0, stream>>>(t0, dw_w, dw_b, dwln_g, dwln_b, x1h);
  // 6) offset proj -> offsets (f32 Mx144)
  {
    EpiParams ep{off_b, nullptr, nullptr, nullptr, nullptr, offsets, nullptr, 0};
    k_gemm_wmma<0><<<gN144, blk, 0, stream>>>(x1h, offwt, MROWS, 144, 256, ep);
  }
  // 7) mask proj -> masks (f32 Mx72), then softmax over P=9
  {
    EpiParams ep{msk_b, nullptr, nullptr, nullptr, nullptr, masks, nullptr, 0};
    k_gemm_wmma<0><<<gN72, blk, 0, stream>>>(x1h, mskwt, MROWS, 72, 256, ep);
  }
  k_softmax9<<<(MROWS * GG + 255) / 256, blk, 0, stream>>>(masks, MROWS * GG);
  // 8) DCNv3 sampling -> dcnouth (bf16)
  k_dcn_sample<<<MROWS, blk, 0, stream>>>(xproj, offsets, masks, dcnouth);
  // 9) output_proj -> dproj (f32)
  {
    EpiParams ep{outp_b, nullptr, nullptr, nullptr, nullptr, dproj, nullptr, 0};
    k_gemm_wmma<0><<<gN256, blk, 0, stream>>>(dcnouth, outwt, MROWS, 256, 256, ep);
  }
  // 10) t1 = t0 + gamma1 * LN(dproj)  (+ bf16 copy)
  k_ln_scale_residual<<<MROWS, blk, 0, stream>>>(t0, dproj, ln1_g, ln1_b, gamma1, t1, t1h);
  // 11) fc1 + GELU -> hh (bf16 Mx1024)
  {
    EpiParams ep{fc1_b, nullptr, nullptr, nullptr, nullptr, nullptr, hh, 1};
    k_gemm_wmma<0><<<gN1024, blk, 0, stream>>>(t1h, fc1wt, MROWS, 1024, 256, ep);
  }
  // 12) fc2 -> hout (f32)
  {
    EpiParams ep{fc2_b, nullptr, nullptr, nullptr, nullptr, hout, nullptr, 0};
    k_gemm_wmma<0><<<gN256, blk, 0, stream>>>(hh, fc2wt, MROWS, 256, 1024, ep);
  }
  // 13) t2 = t1 + gamma2 * LN(hout)
  k_ln_scale_residual<<<MROWS, blk, 0, stream>>>(t1, hout, ln2_g, ln2_b, gamma2, t2, nullptr);
  // 14) out = relu(BN(t2) + x)   (NCHW)
  k_bn_add_relu<<<(total_img + 255) / 256, blk, 0, stream>>>(
      t2, x, bn2_g, bn2_b, bn2_m, bn2_v, (float*)d_out, total_img);
}
